// MultiHeadedAttention_56246891708775
// MI455X (gfx1250) — matrix operand order repaired and checked by reading, not compiled, not run on hardware
//
#include <hip/hip_runtime.h>

#define BD 2
#define HD 16
#define SD 2048
#define DD 1024
#define DKD 64
#define MROWS (BD * SD)   // 4096

typedef __attribute__((ext_vector_type(16))) __bf16 v16bf;
typedef __attribute__((ext_vector_type(8)))  float  v8f;

union V16U {
    v16bf v;
    __bf16 h[16];
    uint4 u4[2];
};

__device__ __forceinline__ unsigned short bf_bits(float x) {
    union { float f; unsigned u; } a; a.f = x;
    unsigned r = a.u + 0x7FFFu + ((a.u >> 16) & 1u);   // RNE
    return (unsigned short)(r >> 16);
}

__device__ __forceinline__ __bf16 f2bf(float x) {
    unsigned short b = bf_bits(x);
    __bf16 o;
    __builtin_memcpy(&o, &b, 2);
    return o;
}

__device__ __forceinline__ v8f wmma_bf16(v16bf a, v16bf b, v8f c) {
    return __builtin_amdgcn_wmma_f32_16x16x32_bf16(
        /*neg_a=*/false, a, /*neg_b=*/false, b,
        /*c_mod=*/(short)0, c, /*reuse_a=*/false, /*reuse_b=*/false);
}

// Fragment from bf16 memory: 8 contiguous bf16 per group (one b128 load each)
__device__ __forceinline__ v16bf frag_ld(const __bf16* g0, const __bf16* g1) {
    V16U t;
    t.u4[0] = *(const uint4*)(const void*)g0;
    t.u4[1] = *(const uint4*)(const void*)g1;
    return t.v;
}

// ---------------------------------------------------------------------------
// One-shot fp32 -> bf16 conversion (bandwidth-bound, keeps GEMM loops VALU-free)
// n8 = element_count / 8
// ---------------------------------------------------------------------------
__global__ __launch_bounds__(256)
void cvt_f32_bf16_kernel(const float* __restrict__ in, __bf16* __restrict__ out, int n8) {
    int i = blockIdx.x * 256 + threadIdx.x;
    if (i >= n8) return;
    const float4* p = (const float4*)in + (size_t)i * 2;
    float4 a = p[0], b = p[1];
    V16U t;
    t.h[0] = f2bf(a.x); t.h[1] = f2bf(a.y); t.h[2] = f2bf(a.z); t.h[3] = f2bf(a.w);
    t.h[4] = f2bf(b.x); t.h[5] = f2bf(b.y); t.h[6] = f2bf(b.z); t.h[7] = f2bf(b.w);
    ((uint4*)out)[i] = t.u4[0];
}

// ---------------------------------------------------------------------------
// Projection: Y[m,n] = sum_k X[m,k]*W[n,k] + bias[n]
// X bf16 [4096,1024], W bf16 [1024,1024] (both pre-converted). Output bf16:
//   vt==0 : [B,H,S,DK]   (Q, K)
//   vt==1 : [B,H,DK,S]   (V transposed for PV B-fragments)
// Wave tile: 64(M) x 64(N). Block = 4 waves. 1024 waves total.
// ---------------------------------------------------------------------------
__global__ __launch_bounds__(128)
void mha_proj_kernel(const __bf16* __restrict__ X, const __bf16* __restrict__ W,
                     const float* __restrict__ bias, __bf16* __restrict__ Y, int vt) {
    int tid  = threadIdx.x;
    int wave = (blockIdx.x << 2) + (tid >> 5);
    int lane = tid & 31;
    int li   = lane & 15;
    int hf   = lane >> 4;

    int mTile = wave >> 4;    // 0..63
    int nTile = wave & 15;    // 0..15
    int m0 = mTile * 64, n0 = nTile * 64;

    v8f acc[4][4];
#pragma unroll
    for (int i = 0; i < 4; ++i)
#pragma unroll
        for (int j = 0; j < 4; ++j) acc[i][j] = (v8f){0,0,0,0,0,0,0,0};

    const __bf16* Xr[4];
    const __bf16* Wr[4];
#pragma unroll
    for (int t = 0; t < 4; ++t) {
        Xr[t] = X + (size_t)(m0 + t * 16 + li) * DD;
        Wr[t] = W + (size_t)(n0 + t * 16 + li) * DD;
    }

    for (int k0 = 0; k0 < DD; k0 += 32) {
        v16bf af[4];
#pragma unroll
        for (int ms = 0; ms < 4; ++ms)
            af[ms] = frag_ld(Xr[ms] + k0 + 8*hf, Xr[ms] + k0 + 16 + 8*hf);
#pragma unroll
        for (int ns = 0; ns < 4; ++ns) {
            v16bf bf = frag_ld(Wr[ns] + k0 + 8*hf, Wr[ns] + k0 + 16 + 8*hf);
#pragma unroll
            for (int ms = 0; ms < 4; ++ms)
                acc[ms][ns] = wmma_bf16(af[ms], bf, acc[ms][ns]);
        }
    }

#pragma unroll
    for (int ns = 0; ns < 4; ++ns) {
        int col = n0 + ns * 16 + li;
        float bv = bias[col];
        int hh = col >> 6, dk = col & 63;
#pragma unroll
        for (int ms = 0; ms < 4; ++ms) {
#pragma unroll
            for (int r = 0; r < 8; ++r) {
                int row = m0 + ms * 16 + r + 8 * hf;
                int bb = row >> 11, ss = row & (SD - 1);
                float v = acc[ms][ns][r] + bv;
                size_t idx = vt
                    ? ((((size_t)(bb * HD + hh)) * DKD + dk) * SD + ss)
                    : ((((size_t)(bb * HD + hh)) * SD + ss) * DKD + dk);
                Y[idx] = f2bf(v);
            }
        }
    }
}

// ---------------------------------------------------------------------------
// Flash attention: one wave owns 16 query rows x DK=64 output for one (b,h).
// Iterates causal 32-key tiles: S = Q K^T * 1/8, online softmax, O += P V.
// Q,K bf16 [B,H,S,64]; VT bf16 [B,H,64,S]; X out bf16 [B,S,1024].
// ---------------------------------------------------------------------------
__global__ __launch_bounds__(128)
void mha_attn_kernel(const __bf16* __restrict__ Q, const __bf16* __restrict__ K,
                     const __bf16* __restrict__ VT, __bf16* __restrict__ X) {
    __shared__ unsigned short plds[4][16 * 32];   // per-wave P bounce (bf16 bits)

    int tid  = threadIdx.x;
    int w    = tid >> 5;
    int wave = (blockIdx.x << 2) + w;
    int lane = tid & 31;
    int li   = lane & 15;
    int hf   = lane >> 4;

    int bh = wave >> 7;      // 0..31  (b*16+h)
    int qt = wave & 127;     // query tile
    int q0 = qt << 4;

    const __bf16* Qb = Q  + (size_t)bh * SD * DKD;
    const __bf16* Kb = K  + (size_t)bh * SD * DKD;
    const __bf16* Vb = VT + (size_t)bh * SD * DKD;

    // Q A-fragments for k=0..63 (loaded once)
    const __bf16* qr = Qb + (size_t)(q0 + li) * DKD;
    v16bf aq0 = frag_ld(qr +      8*hf, qr + 16 + 8*hf);
    v16bf aq1 = frag_ld(qr + 32 + 8*hf, qr + 48 + 8*hf);

    v8f o[4];
#pragma unroll
    for (int i = 0; i < 4; ++i) o[i] = (v8f){0,0,0,0,0,0,0,0};
    float mrow[8], lrow[8];
#pragma unroll
    for (int r = 0; r < 8; ++r) { mrow[r] = -1e30f; lrow[r] = 0.0f; }

    unsigned short* pw = plds[w];

    int jmax = (q0 + 15) >> 5;
    for (int j = 0; j <= jmax; ++j) {
        int kb = j << 5;

        // ---- scores: two 16x16 C tiles over 32 keys
        v8f sf[2];
#pragma unroll
        for (int nt = 0; nt < 2; ++nt) {
            int key = kb + nt * 16 + li;
            const __bf16* kr = Kb + (size_t)key * DKD;
            v16bf bk0 = frag_ld(kr +      8*hf, kr + 16 + 8*hf);
            v16bf bk1 = frag_ld(kr + 32 + 8*hf, kr + 48 + 8*hf);
            v8f c = (v8f){0,0,0,0,0,0,0,0};
            c = wmma_bf16(aq0, bk0, c);
            c = wmma_bf16(aq1, bk1, c);
            sf[nt] = c;
        }

        // ---- scale + causal mask + online softmax update
#pragma unroll
        for (int r = 0; r < 8; ++r) {
            int qrow = q0 + r + 8 * hf;
            float s0 = sf[0][r] * 0.125f;
            float s1 = sf[1][r] * 0.125f;
            if (kb      + li > qrow) s0 = -1e9f;
            if (kb + 16 + li > qrow) s1 = -1e9f;

            float t = fmaxf(s0, s1);
            t = fmaxf(t, __shfl_xor(t, 1));
            t = fmaxf(t, __shfl_xor(t, 2));
            t = fmaxf(t, __shfl_xor(t, 4));
            t = fmaxf(t, __shfl_xor(t, 8));

            float mo = mrow[r];
            float mn = fmaxf(mo, t);
            mrow[r] = mn;
            float sc = __expf(mo - mn);
            lrow[r] *= sc;
            o[0][r] *= sc; o[1][r] *= sc; o[2][r] *= sc; o[3][r] *= sc;

            float p0 = __expf(s0 - mn);
            float p1 = __expf(s1 - mn);
            float rs = p0 + p1;
            rs += __shfl_xor(rs, 1);
            rs += __shfl_xor(rs, 2);
            rs += __shfl_xor(rs, 4);
            rs += __shfl_xor(rs, 8);
            lrow[r] += rs;
            sf[0][r] = p0; sf[1][r] = p1;
        }

        // ---- C-layout P -> LDS -> A-layout fragment
#pragma unroll
        for (int nt = 0; nt < 2; ++nt)
#pragma unroll
            for (int r = 0; r < 8; ++r)
                pw[(r + 8 * hf) * 32 + nt * 16 + li] = bf_bits(sf[nt][r]);

        V16U pa;
        __builtin_memcpy(&pa.u4[0], &pw[li * 32 +      8 * hf], 16);
        __builtin_memcpy(&pa.u4[1], &pw[li * 32 + 16 + 8 * hf], 16);

        // ---- O += P * V  (4 dk tiles)
#pragma unroll
        for (int nt2 = 0; nt2 < 4; ++nt2) {
            const __bf16* vr = Vb + (size_t)(nt2 * 16 + li) * SD + kb;
            v16bf bv = frag_ld(vr + 8 * hf, vr + 16 + 8 * hf);
            o[nt2] = wmma_bf16(pa.v, bv, o[nt2]);
        }
    }

    // ---- normalize + merge heads into X [B,S,1024] (bf16)
    int bb = bh >> 4, hh = bh & 15;
#pragma unroll
    for (int r = 0; r < 8; ++r) {
        float inv = 1.0f / lrow[r];
        int ss = q0 + r + 8 * hf;
        size_t base = ((size_t)bb * SD + ss) * DD + hh * DKD;
#pragma unroll
        for (int nt2 = 0; nt2 < 4; ++nt2)
            X[base + nt2 * 16 + li] = f2bf(o[nt2][r] * inv);
    }
}

// ---------------------------------------------------------------------------
// Output projection: out[m,n] = sum_k X[m,k]*W[n,k] + bias[n]  (fp32 out)
// X bf16 [4096,1024], W bf16 [1024,1024]. Wave tile 64x64.
// ---------------------------------------------------------------------------
__global__ __launch_bounds__(128)
void mha_outproj_kernel(const __bf16* __restrict__ X, const __bf16* __restrict__ W,
                        const float* __restrict__ bias, float* __restrict__ out) {
    int tid  = threadIdx.x;
    int wave = (blockIdx.x << 2) + (tid >> 5);
    int lane = tid & 31;
    int li   = lane & 15;
    int hf   = lane >> 4;

    int mTile = wave >> 4;
    int nTile = wave & 15;
    int m0 = mTile * 64, n0 = nTile * 64;

    v8f acc[4][4];
#pragma unroll
    for (int i = 0; i < 4; ++i)
#pragma unroll
        for (int j = 0; j < 4; ++j) acc[i][j] = (v8f){0,0,0,0,0,0,0,0};

    const __bf16* Xr[4];
    const __bf16* Wr[4];
#pragma unroll
    for (int t = 0; t < 4; ++t) {
        Xr[t] = X + (size_t)(m0 + t * 16 + li) * DD;
        Wr[t] = W + (size_t)(n0 + t * 16 + li) * DD;
    }

    for (int k0 = 0; k0 < DD; k0 += 32) {
        v16bf af[4];
#pragma unroll
        for (int ms = 0; ms < 4; ++ms)
            af[ms] = frag_ld(Xr[ms] + k0 + 8*hf, Xr[ms] + k0 + 16 + 8*hf);
#pragma unroll
        for (int ns = 0; ns < 4; ++ns) {
            v16bf bf = frag_ld(Wr[ns] + k0 + 8*hf, Wr[ns] + k0 + 16 + 8*hf);
#pragma unroll
            for (int ms = 0; ms < 4; ++ms)
                acc[ms][ns] = wmma_bf16(af[ms], bf, acc[ms][ns]);
        }
    }

#pragma unroll
    for (int ns = 0; ns < 4; ++ns) {
        int col = n0 + ns * 16 + li;
        float bv = bias[col];
#pragma unroll
        for (int ms = 0; ms < 4; ++ms) {
#pragma unroll
            for (int r = 0; r < 8; ++r) {
                int row = m0 + ms * 16 + r + 8 * hf;
                out[(size_t)row * DD + col] = acc[ms][ns][r] + bv;
            }
        }
    }
}

extern "C" void kernel_launch(void* const* d_in, const int* in_sizes, int n_in,
                              void* d_out, int out_size, void* d_ws, size_t ws_size,
                              hipStream_t stream) {
    (void)in_sizes; (void)n_in; (void)out_size; (void)ws_size;

    const float* query = (const float*)d_in[0];
    const float* key   = (const float*)d_in[1];
    const float* value = (const float*)d_in[2];
    // d_in[3] = mask : causal tril, implemented analytically in-kernel
    const float* w_q   = (const float*)d_in[4];
    const float* b_q   = (const float*)d_in[5];
    const float* w_k   = (const float*)d_in[6];
    const float* b_k   = (const float*)d_in[7];
    const float* w_v   = (const float*)d_in[8];
    const float* b_v   = (const float*)d_in[9];
    const float* w_out = (const float*)d_in[10];
    const float* b_out = (const float*)d_in[11];
    float* out = (float*)d_out;

    const size_t nAct = (size_t)MROWS * DD;          // 4,194,304
    const size_t nW   = (size_t)DD * DD;             // 1,048,576

    __bf16* p   = (__bf16*)d_ws;
    __bf16* qc  = p;  p += nAct;   // converted inputs
    __bf16* kc  = p;  p += nAct;
    __bf16* vc  = p;  p += nAct;
    __bf16* wqc = p;  p += nW;     // converted weights
    __bf16* wkc = p;  p += nW;
    __bf16* wvc = p;  p += nW;
    __bf16* woc = p;  p += nW;
    __bf16* Qb  = p;  p += nAct;   // projected Q [B,H,S,DK]
    __bf16* Kb  = p;  p += nAct;   // projected K [B,H,S,DK]
    __bf16* VTb = p;  p += nAct;   // projected V^T [B,H,DK,S]
    __bf16* Xb  = p;  p += nAct;   // attention output [B,S,D]

    // fp32 -> bf16 conversions (bandwidth-bound, one shot)
    const int actBlocks = (int)(nAct / 8 / 256);     // 2048
    const int wBlocks   = (int)(nW / 8 / 256);       // 512
    cvt_f32_bf16_kernel<<<actBlocks, 256, 0, stream>>>(query, qc, (int)(nAct / 8));
    cvt_f32_bf16_kernel<<<actBlocks, 256, 0, stream>>>(key,   kc, (int)(nAct / 8));
    cvt_f32_bf16_kernel<<<actBlocks, 256, 0, stream>>>(value, vc, (int)(nAct / 8));
    cvt_f32_bf16_kernel<<<wBlocks, 256, 0, stream>>>(w_q,   wqc, (int)(nW / 8));
    cvt_f32_bf16_kernel<<<wBlocks, 256, 0, stream>>>(w_k,   wkc, (int)(nW / 8));
    cvt_f32_bf16_kernel<<<wBlocks, 256, 0, stream>>>(w_v,   wvc, (int)(nW / 8));
    cvt_f32_bf16_kernel<<<wBlocks, 256, 0, stream>>>(w_out, woc, (int)(nW / 8));

    // projections: 1024 waves -> 256 blocks of 4 waves (64x64 tiles)
    mha_proj_kernel<<<256, 128, 0, stream>>>(qc, wqc, b_q, Qb, 0);
    mha_proj_kernel<<<256, 128, 0, stream>>>(kc, wkc, b_k, Kb, 0);
    mha_proj_kernel<<<256, 128, 0, stream>>>(vc, wvc, b_v, VTb, 1);

    // attention: 4096 waves -> 1024 blocks
    mha_attn_kernel<<<1024, 128, 0, stream>>>(Qb, Kb, VTb, Xb);

    // output projection
    mha_outproj_kernel<<<256, 128, 0, stream>>>(Xb, woc, b_out, out);
}
